// GatedFusion_73641509257452
// MI455X (gfx1250) — matrix operand order repaired and checked by reading, not compiled, not run on hardware
//
#include <hip/hip_runtime.h>
#include <hip/hip_bf16.h>
#include <math.h>

// ---------------------------------------------------------------------------
// GatedFusion on MI455X (gfx1250, wave32, WMMA).
// All matmuls via v_wmma_f32_16x16x32_f16 (f32 accumulate).
// GEMM k-loop is register double-buffered so global fetch of tile k+1
// co-executes with the 8 WMMAs of tile k.
// ---------------------------------------------------------------------------

typedef __attribute__((ext_vector_type(16))) _Float16 v16h;
typedef __attribute__((ext_vector_type(8)))  float    v8f;
typedef __attribute__((ext_vector_type(4)))  float    v4f;

#define BB 8
#define SS 1024
#define DD 1024
#define HH 32
#define DK 32

#define TM 128
#define TN 128
#define TK 32
#define LDA 40   // LDS row pitch in halfs (32 + 8 pad), keeps 16B alignment (80B)
#define LDB 40

union AFrag { v16h v; float f[8]; _Float16 h[16]; };
union BFrag { v16h v; v4f q[2]; };

__device__ __forceinline__ v8f wmma_f32_16x16x32(v16h a, v16h b, v8f c) {
  // (neg_a, A, neg_b, B, c_mod, C, reuse_a, reuse_b)
  return __builtin_amdgcn_wmma_f32_16x16x32_f16(false, a, false, b, (short)0, c,
                                                false, false);
}

// ---------------------------------------------------------------------------
// GEMM: C[M,N] = act( concatK(A1,A2)[M,Ktot] @ W[Ktot,N] + bias ) (+ residual)
// act: 0=none 1=relu 2=sigmoid.  A1/A2 row pitch = K1.  M,N,Ktot multiples of tile.
// ---------------------------------------------------------------------------
__global__ __launch_bounds__(256) void gemm_wmma_kernel(
    const float* __restrict__ A1, const float* __restrict__ A2, int K1, int Ktot,
    const float* __restrict__ W, const float* __restrict__ bias,
    const float* __restrict__ residual, float* __restrict__ C,
    int N, int act)
{
  __shared__ __align__(16) _Float16 lA[TM * LDA];
  __shared__ __align__(16) _Float16 lB[TN * LDB];

  const int tid  = threadIdx.x;
  const int lane = tid & 31;
  const int wv   = tid >> 5;
  const int half = lane >> 4;
  const int ln   = lane & 15;
  const int m0 = blockIdx.y * TM;
  const int n0 = blockIdx.x * TN;
  const int waveM = (wv >> 1) * 32;   // 4 waves along M
  const int waveN = (wv & 1) * 64;    // 2 waves along N

  v8f acc[2][4] = {};

  // ---- register double-buffer: prologue fetch of tile k0 = 0 ----------
  float ra[16], rb[16];
#pragma unroll
  for (int j = 0; j < 16; ++j) {
    int i = tid + 256 * j;
    {
      int r = i >> 5, c = i & 31;                 // A: 128 x 32
      ra[j] = (c < K1) ? A1[(size_t)(m0 + r) * K1 + c]
                       : A2[(size_t)(m0 + r) * K1 + (c - K1)];
    }
    {
      int c = i & 127, r = i >> 7;                // W: 32 x 128
      rb[j] = W[(size_t)r * N + (n0 + c)];
    }
  }

  for (int k0 = 0; k0 < Ktot; k0 += TK) {
    // commit staged registers to LDS (A row-major f16, W transposed f16)
#pragma unroll
    for (int j = 0; j < 16; ++j) {
      int i = tid + 256 * j;
      lA[(i >> 5) * LDA + (i & 31)]  = (_Float16)ra[j];
      lB[(i & 127) * LDB + (i >> 7)] = (_Float16)rb[j];
    }
    __syncthreads();

    // issue global fetch of tile k0+TK; completes under the WMMAs below
    if (k0 + TK < Ktot) {
      const int k1 = k0 + TK;
      if (k0 + 2 * TK < Ktot)   // light L2 prefetch two tiles ahead
        __builtin_prefetch(&W[(size_t)(k0 + 2 * TK + (tid >> 3)) * N + n0 + ((tid & 7) << 4)], 0, 1);
#pragma unroll
      for (int j = 0; j < 16; ++j) {
        int i = tid + 256 * j;
        int r = i >> 5, c = i & 31, gk = k1 + c;
        ra[j] = (gk < K1) ? A1[(size_t)(m0 + r) * K1 + gk]
                          : A2[(size_t)(m0 + r) * K1 + (gk - K1)];
        int cb = i & 127, rk = i >> 7;
        rb[j] = W[(size_t)(k1 + rk) * N + (n0 + cb)];
      }
    }

    // A fragments: lane row = waveM + mt*16 + ln; K = half*8+e | 16+half*8+(e-8)
    AFrag af[2];
#pragma unroll
    for (int mt = 0; mt < 2; ++mt) {
      const _Float16* base = &lA[(waveM + mt * 16 + ln) * LDA];
#pragma unroll
      for (int j = 0; j < 4; ++j) {
        af[mt].f[j]     = *(const float*)(base + half * 8 + 2 * j);
        af[mt].f[4 + j] = *(const float*)(base + 16 + half * 8 + 2 * j);
      }
    }
    // B fragments: lane col = waveN + nt*16 + ln; K = half*8+e | 16+half*8+(e-8)
    BFrag bf[4];
#pragma unroll
    for (int nt = 0; nt < 4; ++nt) {
      const _Float16* base = &lB[(waveN + nt * 16 + ln) * LDB + half * 8];
      bf[nt].q[0] = *(const v4f*)(base);
      bf[nt].q[1] = *(const v4f*)(base + 16);
    }
#pragma unroll
    for (int mt = 0; mt < 2; ++mt)
#pragma unroll
      for (int nt = 0; nt < 4; ++nt)
        acc[mt][nt] = wmma_f32_16x16x32(af[mt].v, bf[nt].v, acc[mt][nt]);
    __syncthreads();
  }

  // epilogue: C/D layout row = 8*half + r, col = ln
#pragma unroll
  for (int mt = 0; mt < 2; ++mt) {
#pragma unroll
    for (int nt = 0; nt < 4; ++nt) {
      int col = n0 + waveN + nt * 16 + ln;
      float bv = bias[col];
#pragma unroll
      for (int r = 0; r < 8; ++r) {
        int row = m0 + waveM + mt * 16 + half * 8 + r;
        float v = acc[mt][nt][r] + bv;
        if (act == 1)      v = fmaxf(v, 0.0f);
        else if (act == 2) v = __builtin_amdgcn_rcpf(1.0f + __expf(-v)); // fast sigmoid
        if (residual)      v += residual[(size_t)row * N + col];
        C[(size_t)row * N + col] = v;
      }
    }
  }
}

// ---------------------------------------------------------------------------
// Flash attention: O = softmax(Q Kt / sqrt(dk)) V, per (b,h). dk==32==WMMA K.
// Block = 128 threads = 4 waves; each wave owns 16 q-rows; grid (S/64, H, B).
// ---------------------------------------------------------------------------
__global__ __launch_bounds__(128) void attn_kernel(
    const float* __restrict__ Q, const float* __restrict__ Kg,
    const float* __restrict__ Vg, float* __restrict__ O)
{
  __shared__ __align__(16) _Float16 sK[32 * 40];       // [key][dk]
  __shared__ __align__(16) _Float16 sVt[32 * 40];      // [dk][key]
  __shared__ __align__(16) _Float16 sP[4][16 * 40];    // per-wave P (16 x 32)

  const int tid  = threadIdx.x;
  const int lane = tid & 31, wv = tid >> 5;
  const int half = lane >> 4, ln = lane & 15;
  const int h = blockIdx.y, b = blockIdx.z;
  const int q0 = blockIdx.x * 64 + wv * 16;

  // Q fragment (A layout), pre-scaled by 1/sqrt(dk)
  AFrag qa;
  {
    const float scale = 0.17677669529663687f;
    const float* qp = Q + ((size_t)(b * SS + q0 + ln)) * DD + h * DK;
#pragma unroll
    for (int e = 0; e < 16; ++e) {
      int kd = (e < 8) ? (half * 8 + e) : (16 + half * 8 + (e - 8));
      qa.h[e] = (_Float16)(qp[kd] * scale);
    }
  }

  float mrow[8], lrow[8];
#pragma unroll
  for (int r = 0; r < 8; ++r) { mrow[r] = -1e30f; lrow[r] = 0.0f; }
  v8f o0 = {}, o1 = {};
  v8f zero = {};

  for (int j0 = 0; j0 < SS; j0 += 32) {
    // cooperative stage of 32 keys: K row-major, V transposed
#pragma unroll
    for (int s = 0; s < 8; ++s) {
      int i = tid + 128 * s;
      int key = i >> 5, d = i & 31;
      size_t g = ((size_t)(b * SS + j0 + key)) * DD + h * DK + d;
      sK[key * 40 + d]  = (_Float16)Kg[g];
      sVt[d * 40 + key] = (_Float16)Vg[g];
    }
    __syncthreads();

    // K^T fragments (B layout) and V fragments (B layout) -> registers
    BFrag kb0, kb1, vb0, vb1;
    {
      const _Float16* p0 = &sK[ln * 40 + half * 8];
      const _Float16* p1 = &sK[(16 + ln) * 40 + half * 8];
      kb0.q[0] = *(const v4f*)p0; kb0.q[1] = *(const v4f*)(p0 + 16);
      kb1.q[0] = *(const v4f*)p1; kb1.q[1] = *(const v4f*)(p1 + 16);
      const _Float16* p2 = &sVt[ln * 40 + half * 8];
      const _Float16* p3 = &sVt[(16 + ln) * 40 + half * 8];
      vb0.q[0] = *(const v4f*)p2; vb0.q[1] = *(const v4f*)(p2 + 16);
      vb1.q[0] = *(const v4f*)p3; vb1.q[1] = *(const v4f*)(p3 + 16);
    }

    v8f sc0 = wmma_f32_16x16x32(qa.v, kb0.v, zero);   // keys j0..j0+15
    v8f sc1 = wmma_f32_16x16x32(qa.v, kb1.v, zero);   // keys j0+16..j0+31

    // online softmax (rows 8*half+r; 16 lanes per row -> shfl_xor < 16)
#pragma unroll
    for (int r = 0; r < 8; ++r) {
      float cm = fmaxf(sc0[r], sc1[r]);
#pragma unroll
      for (int off = 8; off >= 1; off >>= 1)
        cm = fmaxf(cm, __shfl_xor(cm, off, 32));
      float mn = fmaxf(mrow[r], cm);
      float alpha = __expf(mrow[r] - mn);
      float p0 = __expf(sc0[r] - mn);
      float p1 = __expf(sc1[r] - mn);
      float rs = p0 + p1;
#pragma unroll
      for (int off = 8; off >= 1; off >>= 1)
        rs += __shfl_xor(rs, off, 32);
      lrow[r] = lrow[r] * alpha + rs;
      mrow[r] = mn;
      o0[r] *= alpha;
      o1[r] *= alpha;
      sP[wv][(half * 8 + r) * 40 + ln]      = (_Float16)p0;
      sP[wv][(half * 8 + r) * 40 + 16 + ln] = (_Float16)p1;
    }
    __syncthreads();   // sK/sVt reads retired; sP visible

    // P fragment (A layout, wave-private LDS region)
    AFrag pf;
    {
      const _Float16* base = &sP[wv][ln * 40];
#pragma unroll
      for (int j = 0; j < 4; ++j) {
        pf.f[j]     = *(const float*)(base + half * 8 + 2 * j);
        pf.f[4 + j] = *(const float*)(base + 16 + half * 8 + 2 * j);
      }
    }
    o0 = wmma_f32_16x16x32(pf.v, vb0.v, o0);   // dk cols 0..15
    o1 = wmma_f32_16x16x32(pf.v, vb1.v, o1);   // dk cols 16..31
  }

#pragma unroll
  for (int r = 0; r < 8; ++r) {
    float inv = __builtin_amdgcn_rcpf(lrow[r]);
    size_t base = ((size_t)(b * SS + q0 + half * 8 + r)) * DD + h * DK;
    O[base + ln]      = o0[r] * inv;
    O[base + 16 + ln] = o1[r] * inv;
  }
}

// ---------------------------------------------------------------------------
// In-place LayerNorm over last dim (D=1024); one block per row.
// ---------------------------------------------------------------------------
__global__ __launch_bounds__(256) void ln_kernel(float* __restrict__ x,
    const float* __restrict__ g, const float* __restrict__ beta)
{
  __shared__ float sb[256];
  const int tid = threadIdx.x;
  float* row = x + (size_t)blockIdx.x * DD;
  float v[4];
  float s = 0.0f;
#pragma unroll
  for (int k = 0; k < 4; ++k) { v[k] = row[tid + 256 * k]; s += v[k]; }
  sb[tid] = s; __syncthreads();
  for (int st = 128; st > 0; st >>= 1) {
    if (tid < st) sb[tid] += sb[tid + st];
    __syncthreads();
  }
  float mean = sb[0] * (1.0f / DD);
  __syncthreads();
  float s2 = 0.0f;
#pragma unroll
  for (int k = 0; k < 4; ++k) { float d = v[k] - mean; s2 += d * d; }
  sb[tid] = s2; __syncthreads();
  for (int st = 128; st > 0; st >>= 1) {
    if (tid < st) sb[tid] += sb[tid + st];
    __syncthreads();
  }
  float rstd = rsqrtf(sb[0] * (1.0f / DD) + 1e-5f);
#pragma unroll
  for (int k = 0; k < 4; ++k) {
    int c = tid + 256 * k;
    row[c] = (v[k] - mean) * rstd * g[c] + beta[c];
  }
}

// ---------------------------------------------------------------------------
// Elementwise helpers
// ---------------------------------------------------------------------------
__global__ void add_kernel(float* __restrict__ o, const float* __restrict__ a,
                           const float* __restrict__ b, size_t n) {
  size_t i = (size_t)blockIdx.x * blockDim.x + threadIdx.x;
  size_t st = (size_t)gridDim.x * blockDim.x;
  for (; i < n; i += st) o[i] = a[i] + b[i];
}

__global__ void gatefuse_kernel(float* __restrict__ o, const float* __restrict__ gate,
                                const float* __restrict__ a, const float* __restrict__ b,
                                size_t n) {
  size_t i = (size_t)blockIdx.x * blockDim.x + threadIdx.x;
  size_t st = (size_t)gridDim.x * blockDim.x;
  for (; i < n; i += st) {
    float g = gate[i];
    o[i] = g * a[i] + (1.0f - g) * b[i];
  }
}

// ---------------------------------------------------------------------------
// Host orchestration
// ---------------------------------------------------------------------------
extern "C" void kernel_launch(void* const* d_in, const int* in_sizes, int n_in,
                              void* d_out, int out_size, void* d_ws, size_t ws_size,
                              hipStream_t stream) {
  const int M = BB * SS;          // 8192 token rows
  const int N = DD;               // 1024
  const size_t E = (size_t)M * DD;

  const float* poi_emb  = (const float*)d_in[0];
  const float* svi_emb  = (const float*)d_in[1];
  const float* poly_emb = (const float*)d_in[2];
  const float* Wq = (const float*)d_in[3];  const float* bq = (const float*)d_in[4];
  const float* Wk = (const float*)d_in[5];  const float* bk = (const float*)d_in[6];
  const float* Wv = (const float*)d_in[7];  const float* bv = (const float*)d_in[8];
  const float* Wo = (const float*)d_in[9];  const float* bo = (const float*)d_in[10];
  const float* Wpoi = (const float*)d_in[11];  const float* bpoi = (const float*)d_in[12];
  const float* gpoi = (const float*)d_in[13];  const float* bepoi = (const float*)d_in[14];
  const float* Wsvi = (const float*)d_in[15];  const float* bsvi = (const float*)d_in[16];
  const float* gsvi = (const float*)d_in[17];  const float* besvi = (const float*)d_in[18];
  const float* Wpoly = (const float*)d_in[19]; const float* bpoly = (const float*)d_in[20];
  const float* gpoly = (const float*)d_in[21]; const float* bepoly = (const float*)d_in[22];
  const float* Wg = (const float*)d_in[23]; const float* bg = (const float*)d_in[24];
  const float* Wf = (const float*)d_in[25]; const float* bf = (const float*)d_in[26];
  const float* Wt = (const float*)d_in[27]; const float* bt = (const float*)d_in[28];

  float* s0 = (float*)d_ws;
  float* s1 = s0 + E;
  float* s2 = s1 + E;
  float* s3 = s2 + E;
  float* s4 = s3 + E;
  float* s5 = s4 + E;
  float* s6 = s5 + E;

  const dim3 gGrid(N / TN, M / TM);       // (8, 64)
  const dim3 aGrid(SS / 64, HH, BB);      // (16, 32, 8)
  const size_t n = E;
  const int ewBlocks = 4096;

  auto gemm = [&](const float* A, const float* W, const float* bias, float* C,
                  int act, const float* res) {
    gemm_wmma_kernel<<<gGrid, 256, 0, stream>>>(A, A, DD, DD, W, bias, res, C, N, act);
  };
  auto gemm2 = [&](const float* A1, const float* A2, const float* W,
                   const float* bias, float* C, int act) {
    gemm_wmma_kernel<<<gGrid, 256, 0, stream>>>(A1, A2, DD, 2 * DD, W, bias, nullptr, C, N, act);
  };
  auto ln = [&](float* x, const float* g, const float* be) {
    ln_kernel<<<M, 256, 0, stream>>>(x, g, be);
  };
  auto attn = [&](const float* Q, const float* K, const float* V, float* O) {
    attn_kernel<<<aGrid, 128, 0, stream>>>(Q, K, V, O);
  };

  // ---- stage 1: input transforms --------------------------------------
  gemm(poi_emb, Wpoi, bpoi, s0, 1, nullptr);  ln(s0, gpoi, bepoi);    // poi -> s0
  gemm(svi_emb, Wsvi, bsvi, s1, 1, nullptr);  ln(s1, gsvi, besvi);    // svi -> s1

  // ---- mha #1: svi_feat = mha(svi, poi, poi) --------------------------
  gemm(s1, Wq, bq, s2, 0, nullptr);
  gemm(s0, Wk, bk, s3, 0, nullptr);
  gemm(s0, Wv, bv, s4, 0, nullptr);
  attn(s2, s3, s4, s5);
  gemm(s5, Wo, bo, s2, 0, nullptr);                                   // svi_feat0 -> s2

  // ---- mha #2: poi_feat = mha(poi, svi_feat0, svi_feat0) + poi --------
  gemm(s0, Wq, bq, s3, 0, nullptr);
  gemm(s2, Wk, bk, s4, 0, nullptr);
  gemm(s2, Wv, bv, s5, 0, nullptr);
  attn(s3, s4, s5, s6);
  gemm(s6, Wo, bo, s3, 0, s0);                                        // poi_feat -> s3
  add_kernel<<<ewBlocks, 256, 0, stream>>>(s4, s2, s1, n);            // svi_feat -> s4

  // ---- gate / fusion / filtration / out -------------------------------
  gemm2(s3, s4, Wg, bg, s0, 2);                                       // gate -> s0
  gatefuse_kernel<<<ewBlocks, 256, 0, stream>>>(s1, s0, s3, s4, n);   // svi_fusion -> s1
  gemm2(s3, s1, Wf, bf, s2, 1);                                       // filt -> s2
  gemm(s2, Wt, bt, s0, 0, nullptr);                                   // out -> s0

  // ---- stage 2 (num==3): poly branch ----------------------------------
  gemm(poly_emb, Wpoly, bpoly, s1, 1, nullptr); ln(s1, gpoly, bepoly); // poly -> s1

  // mha #3: poly_feat0 = mha(poly, out, out)
  gemm(s1, Wq, bq, s2, 0, nullptr);
  gemm(s0, Wk, bk, s3, 0, nullptr);
  gemm(s0, Wv, bv, s4, 0, nullptr);
  attn(s2, s3, s4, s5);
  gemm(s5, Wo, bo, s2, 0, nullptr);                                   // poly_feat0 -> s2

  // mha #4: poi_feat2 = mha(out, poly_feat0, poly_feat0) + out
  gemm(s0, Wq, bq, s3, 0, nullptr);
  gemm(s2, Wk, bk, s4, 0, nullptr);
  gemm(s2, Wv, bv, s5, 0, nullptr);
  attn(s3, s4, s5, s6);
  gemm(s6, Wo, bo, s3, 0, s0);                                        // poi_feat2 -> s3
  add_kernel<<<ewBlocks, 256, 0, stream>>>(s4, s2, s1, n);            // poly_feat -> s4

  gemm2(s3, s4, Wg, bg, s0, 2);                                       // gate2 -> s0
  gatefuse_kernel<<<ewBlocks, 256, 0, stream>>>(s1, s0, s3, s4, n);   // poly_fusion -> s1
  gemm2(s3, s1, Wf, bf, s2, 1);                                       // filt2 -> s2
  gemm(s2, Wt, bt, (float*)d_out, 0, nullptr);                        // final output
}
